// MultiHeadAttention__40870908788989
// MI455X (gfx1250) — matrix operand order repaired and checked by reading, not compiled, not run on hardware
//
#include <hip/hip_runtime.h>
#include <cstdint>
#include <math.h>

typedef __attribute__((ext_vector_type(2)))  float    v2f;
typedef __attribute__((ext_vector_type(8)))  float    v8f;
typedef __attribute__((ext_vector_type(2)))  __bf16   v2bf;
typedef __attribute__((ext_vector_type(16))) __bf16   v16bf;
typedef __attribute__((ext_vector_type(4)))  unsigned u32x4;
typedef __attribute__((ext_vector_type(8)))  unsigned u32x8;

#define S_LEN 2048
#define DHEAD 64
#define NHEAD 8
#define NBATCH 4

#if __has_builtin(__builtin_amdgcn_wmma_f32_16x16x4_f32)
#define HAVE_WMMA_F32 1
#else
#define HAVE_WMMA_F32 0
#endif

// two floats -> packed bf16x2 via native converts (lowers to v_cvt_pk_bf16_f32)
__device__ __forceinline__ unsigned pack_bf16(float lo, float hi) {
    union { v2bf v; unsigned u; } c;
    c.v.x = (__bf16)lo;
    c.v.y = (__bf16)hi;
    return c.u;
}

__device__ __forceinline__ unsigned short cvt_bf16(float x) {
    union { __bf16 b; unsigned short u; } c;
    c.b = (__bf16)x;
    return c.u;
}

// Build a v16bf A/B fragment from two 16-byte chunks
__device__ __forceinline__ v16bf load_bf16x16(const unsigned short* p0,
                                              const unsigned short* p1) {
    union { v16bf v; float4 q[2]; } u;
    u.q[0] = *(const float4*)p0;
    u.q[1] = *(const float4*)p1;
    return u.v;
}

// ---------------------------------------------------------------------------
// Tensor Data Mover helpers (CDNA5 ISA ch.8): DMA a 2D tile of 8-byte
// elements (nrows x row_qw, row pitch stride_qw) from global into LDS.
// D# group0/group1 built per ISA §8.3/§8.4; issued per-wave via inline asm
// (portable across toolchains with differing builtin arity).
// ---------------------------------------------------------------------------
__device__ __forceinline__ unsigned lds_offset(const void* p) {
    return (unsigned)(uintptr_t)p;   // generic LDS pointer: low 32 bits = LDS byte offset
}

__device__ __forceinline__ void tdm_load_tile(unsigned lds_addr,
                                              const void* gaddr,
                                              unsigned row_qw, unsigned nrows,
                                              unsigned stride_qw) {
    unsigned long long ga = (unsigned long long)(uintptr_t)gaddr;
    u32x4 g0;
    g0[0] = 1u;                                   // count=1 (valid user D#)
    g0[1] = lds_addr;                             // lds_addr (bytes)
    g0[2] = (unsigned)ga;                         // global_addr[31:0]
    g0[3] = (unsigned)((ga >> 32) & 0x01FFFFFFu)  // global_addr[56:32]
            | (2u << 30);                         // type = 2 ("image")
    u32x8 g1;
    g1[0] = 3u << 16;        // workgroup_mask=0, data_size=3 (8 bytes)
    g1[1] = 0x4000u << 16;   // tensor_dim0[15:0]=0x4000 (big: no OOB clamp)
    g1[2] = 0x4000u << 16;   // tensor_dim0 hi=0, tensor_dim1[15:0]=0x4000
    g1[3] = row_qw << 16;    // tensor_dim1 hi=0, tile_dim0=row_qw
    g1[4] = nrows;           // tile_dim1=nrows (0 => single row), tile_dim2=0
    g1[5] = stride_qw;       // tensor_dim0_stride[31:0]
    g1[6] = 0u;              // stride hi / tensor_dim1_stride lo
    g1[7] = 0u;              // tensor_dim1_stride hi
    asm volatile("tensor_load_to_lds %0, %1" :: "s"(g0), "s"(g1) : "memory");
}

__device__ __forceinline__ void wait_tensorcnt0() {
#if __has_builtin(__builtin_amdgcn_s_wait_tensorcnt)
    __builtin_amdgcn_s_wait_tensorcnt(0);
#else
    asm volatile("s_wait_tensorcnt 0x0" ::: "memory");
#endif
}

// ---------------------------------------------------------------------------
// Kernel 0: Wt_bf[d][k] = bf16(W[k][d])   (W: 512x64 -> Wt: 64x512 bf16)
// ---------------------------------------------------------------------------
__global__ void wtrans_kernel(const float* __restrict__ W,
                              unsigned short* __restrict__ Wt) {
    for (int i = threadIdx.x + blockIdx.x * blockDim.x; i < 64 * 256;
         i += blockDim.x * gridDim.x) {
        const int d = i & 63;
        const int k = (i >> 6) * 2;
        unsigned p = pack_bf16(W[k * 64 + d], W[(k + 1) * 64 + d]);
        *(unsigned*)&Wt[d * 512 + k] = p;
    }
}

// ---------------------------------------------------------------------------
// Kernel 1 helper: one 64x2048 projection for one (b,h) with f32 WMMA.
// ---------------------------------------------------------------------------
template <int MAT>
__device__ __forceinline__ void proj_one(
    const float* __restrict__ x, const float* __restrict__ Wm,
    float* __restrict__ O, unsigned short* __restrict__ Sg,
    int b, int bh, int n, int l16, int lhalf) {
    v8f C[4] = {v8f{0}, v8f{0}, v8f{0}, v8f{0}};

#if HAVE_WMMA_F32
    for (int k0 = 0; k0 < DHEAD; k0 += 4) {
        const int kb = k0 + lhalf * 2;   // A frag: lanes 0-15 K={0,1}, 16-31 K={2,3}
        v2f bv;
        bv.x = x[(size_t)(b * DHEAD + kb) * S_LEN + n];
        bv.y = x[(size_t)(b * DHEAD + kb + 1) * S_LEN + n];
#pragma unroll
        for (int rt = 0; rt < 4; ++rt) {
            v2f av = *(const v2f*)(Wm + (rt * 16 + l16) * DHEAD + kb);
            C[rt] = __builtin_amdgcn_wmma_f32_16x16x4_f32(
                false, av, false, bv, (short)0, C[rt], false, false);
        }
    }
#else
    for (int k = 0; k < DHEAD; ++k) {
        const float xv = x[(size_t)(b * DHEAD + k) * S_LEN + n];
#pragma unroll
        for (int rt = 0; rt < 4; ++rt)
#pragma unroll
            for (int e = 0; e < 8; ++e)
                C[rt][e] += Wm[(rt * 16 + e + lhalf * 8) * DHEAD + k] * xv;
    }
#endif

#pragma unroll
    for (int rt = 0; rt < 4; ++rt) {
#pragma unroll
        for (int e = 0; e < 8; ++e) {
            const int d = rt * 16 + e + lhalf * 8;
            O[(size_t)d * S_LEN + n] = C[rt][e];
        }
        if (MAT == 0) {          // Qt: s-major, fold in 1/sqrt(64)
#pragma unroll
            for (int e = 0; e < 8; e += 2) {
                const int d = rt * 16 + e + lhalf * 8;
                *(unsigned*)&Sg[((size_t)bh * S_LEN + n) * DHEAD + d] =
                    pack_bf16(C[rt][e] * 0.125f, C[rt][e + 1] * 0.125f);
            }
        } else if (MAT == 1) {   // Kt: s-major
#pragma unroll
            for (int e = 0; e < 8; e += 2) {
                const int d = rt * 16 + e + lhalf * 8;
                *(unsigned*)&Sg[((size_t)bh * S_LEN + n) * DHEAD + d] =
                    pack_bf16(C[rt][e], C[rt][e + 1]);
            }
        } else {                 // Vb: d-major
#pragma unroll
            for (int e = 0; e < 8; ++e) {
                const int d = rt * 16 + e + lhalf * 8;
                Sg[((size_t)bh * DHEAD + d) * S_LEN + n] = cvt_bf16(C[rt][e]);
            }
        }
    }
}

// grid: (B*H, S/128), block 256 (8 waves, each owns 16 columns)
__global__ __launch_bounds__(256) void qkv_proj_kernel(
    const float* __restrict__ x,
    const float* __restrict__ Wq, const float* __restrict__ Wk,
    const float* __restrict__ Wv,
    float* __restrict__ Qo, float* __restrict__ Ko, float* __restrict__ Vo,
    unsigned short* __restrict__ Qt, unsigned short* __restrict__ Kt,
    unsigned short* __restrict__ Vb) {
    const int bh    = blockIdx.x;
    const int b     = bh >> 3;
    const int h     = bh & 7;
    const int wave  = threadIdx.x >> 5;
    const int lane  = threadIdx.x & 31;
    const int l16   = lane & 15;
    const int lhalf = lane >> 4;
    const int n     = blockIdx.y * 128 + wave * 16 + l16;   // global column s

    proj_one<0>(x, Wq + h * DHEAD * DHEAD, Qo + (size_t)bh * DHEAD * S_LEN, Qt,
                b, bh, n, l16, lhalf);
    proj_one<1>(x, Wk + h * DHEAD * DHEAD, Ko + (size_t)bh * DHEAD * S_LEN, Kt,
                b, bh, n, l16, lhalf);
    proj_one<2>(x, Wv + h * DHEAD * DHEAD, Vo + (size_t)bh * DHEAD * S_LEN, Vb,
                b, bh, n, l16, lhalf);
}

// ---------------------------------------------------------------------------
// Kernel 2: fused flash attention per (b,h, 64 t-columns).
// Q/V tiles double-buffered into LDS via TDM (tensor_load_to_lds, TENSORcnt);
// L[s,t] = sum_d Qt[s,d]*Kt[t,d]; online softmax over s (4 threads/column);
// O[d,t] += Vb[d,s] @ P[s,t].  bf16 WMMA 16x16x32.
// grid: (B*H, S/64), block 256 (8 waves: row group r=w&3, col pair c=w>>2)
// ---------------------------------------------------------------------------
__global__ __launch_bounds__(256) void attn_kernel(
    const unsigned short* __restrict__ Qt, const unsigned short* __restrict__ Kt,
    const unsigned short* __restrict__ Vb, unsigned short* __restrict__ outT) {
    __shared__ float          Lsm[64 * 68];     // logits, s-major
    __shared__ unsigned short Ptm[64 * 72];     // P, t-major bf16
    __shared__ float          fac[64];          // per-column rescale / 1/l
    __shared__ float          mnS[64];          // per-column running max
    __shared__ float          red[4 * 64];      // partial max/sum reduction
    __shared__ alignas(16) unsigned short Qs[2][64 * 64];  // TDM-staged Q tiles
    __shared__ alignas(16) unsigned short Vs[2][64 * 64];  // TDM-staged V tiles

    const int bh    = blockIdx.x;
    const int b     = bh >> 3;
    const int h     = bh & 7;
    const int t0    = blockIdx.y * 64;
    const int tid   = threadIdx.x;
    const int wave  = tid >> 5;
    const int lane  = tid & 31;
    const int l16   = lane & 15;
    const int lhalf = lane >> 4;
    const int r     = wave & 3;            // s-row group (L) / d-row group (O)
    const int cb    = (wave >> 2) * 2;     // first of two 16-col tiles
    const int tcol  = tid & 63;            // softmax: column owned
    const int tq    = tid >> 6;            // softmax: s-quarter owned

    // Preload K B-fragments (constant over s): per-lane 16 contiguous d
    v16bf Bk[2][2];
#pragma unroll
    for (int i = 0; i < 2; ++i) {
        const int t = t0 + (cb + i) * 16 + l16;
        const unsigned short* rowp = Kt + ((size_t)bh * S_LEN + t) * DHEAD;
#pragma unroll
        for (int ks = 0; ks < 2; ++ks) {
            const unsigned short* p = rowp + ks * 32 + lhalf * 16;
            Bk[i][ks] = load_bf16x16(p, p + 8);
        }
    }

    v8f  O[2] = {v8f{0}, v8f{0}};
    float mrun = -INFINITY, lrun = 0.f;

    const unsigned short* gQ = Qt + (size_t)bh * S_LEN * DHEAD;   // + s*64
    const unsigned short* gV = Vb + (size_t)bh * DHEAD * S_LEN;   // rows: stride S

    // Kick off TDM for the first s-block (one wave issues; EXEC-independent)
    if (wave == 0) {
        tdm_load_tile(lds_offset(&Qs[0][0]), gQ, 1024, 0, 1024);  // 8KB contiguous
        tdm_load_tile(lds_offset(&Vs[0][0]), gV, 16, 64, 512);    // 64 rows x 128B
    }

    int buf = 0;
    for (int s0 = 0; s0 < S_LEN; s0 += 64) {
        if (wave == 0) wait_tensorcnt0();    // current buffer landed
        __syncthreads();
        // overlap: issue DMA for the next s-block into the other buffer
        if (wave == 0 && s0 + 64 < S_LEN) {
            tdm_load_tile(lds_offset(&Qs[buf ^ 1][0]),
                          gQ + (size_t)(s0 + 64) * DHEAD, 1024, 0, 1024);
            tdm_load_tile(lds_offset(&Vs[buf ^ 1][0]),
                          gV + (s0 + 64), 16, 64, 512);
        }

        // ---- L tiles: A = staged Q rows (s), B = preloaded K fragments ----
        {
            const unsigned short* arow = &Qs[buf][(r * 16 + l16) * DHEAD];
            v8f Lc[2] = {v8f{0}, v8f{0}};
#pragma unroll
            for (int ks = 0; ks < 2; ++ks) {
                const unsigned short* pa = arow + ks * 32 + lhalf * 16;
                v16bf A = load_bf16x16(pa, pa + 8);
#pragma unroll
                for (int i = 0; i < 2; ++i)
                    Lc[i] = __builtin_amdgcn_wmma_f32_16x16x32_bf16(
                        false, A, false, Bk[i][ks], (short)0, Lc[i], false, false);
            }
#pragma unroll
            for (int i = 0; i < 2; ++i) {
                const int tl = (cb + i) * 16 + l16;
#pragma unroll
                for (int e = 0; e < 8; ++e) {
                    const int sl = r * 16 + e + lhalf * 8;
                    Lsm[sl * 68 + tl] = Lc[i][e];
                }
            }
        }
        __syncthreads();

        // ---- softmax phase A: partial max (4 threads per column) ----
        {
            float bm = -INFINITY;
            const int sb = tq * 16;
#pragma unroll
            for (int s = 0; s < 16; ++s)
                bm = fmaxf(bm, Lsm[(sb + s) * 68 + tcol]);
            red[tq * 64 + tcol] = bm;
        }
        __syncthreads();
        if (tid < 64) {
            float bm = fmaxf(fmaxf(red[tcol], red[64 + tcol]),
                             fmaxf(red[128 + tcol], red[192 + tcol]));
            const float mnew = fmaxf(mrun, bm);
            fac[tcol] = __expf(mrun - mnew);
            mnS[tcol] = mnew;
            mrun      = mnew;
        }
        __syncthreads();

        // ---- softmax phase B: exp + packed bf16 P + partial sum ----
        {
            const float mnew = mnS[tcol];
            const int sb = tq * 16;
            float sum = 0.f;
#pragma unroll
            for (int s = 0; s < 16; s += 2) {
                const float p0 = __expf(Lsm[(sb + s) * 68 + tcol] - mnew);
                const float p1 = __expf(Lsm[(sb + s + 1) * 68 + tcol] - mnew);
                *(unsigned*)&Ptm[tcol * 72 + sb + s] = pack_bf16(p0, p1);
                sum += p0 + p1;
            }
            red[tq * 64 + tcol] = sum;
        }
        __syncthreads();
        if (tid < 64)
            lrun = lrun * fac[tcol] +
                   (red[tcol] + red[64 + tcol]) + (red[128 + tcol] + red[192 + tcol]);

        // ---- rescale O, then O += V @ P (both operands from LDS) ----
        {
            const float f0 = fac[(cb + 0) * 16 + l16];
            const float f1 = fac[(cb + 1) * 16 + l16];
#pragma unroll
            for (int e = 0; e < 8; ++e) { O[0][e] *= f0; O[1][e] *= f1; }

            const unsigned short* vrow = &Vs[buf][(r * 16 + l16) * 64];
#pragma unroll
            for (int ks = 0; ks < 2; ++ks) {
                const unsigned short* pa = vrow + ks * 32 + lhalf * 16;
                v16bf A = load_bf16x16(pa, pa + 8);
#pragma unroll
                for (int i = 0; i < 2; ++i) {
                    const unsigned short* pb =
                        &Ptm[((cb + i) * 16 + l16) * 72 + ks * 32 + lhalf * 16];
                    v16bf B = load_bf16x16(pb, pb + 8);
                    O[i] = __builtin_amdgcn_wmma_f32_16x16x32_bf16(
                        false, A, false, B, (short)0, O[i], false, false);
                }
            }
        }
        __syncthreads();
        buf ^= 1;
    }

    if (tid < 64) fac[tid] = 1.0f / lrun;
    __syncthreads();

    // outT[b][t][h*64+d] (t-major bf16 => contiguous B-fragments for kernel 3)
#pragma unroll
    for (int i = 0; i < 2; ++i) {
        const int tl  = (cb + i) * 16 + l16;
        const int t   = t0 + tl;
        const float f = fac[tl];
        unsigned short* orow = outT + ((size_t)b * S_LEN + t) * 512 + h * 64;
#pragma unroll
        for (int e = 0; e < 8; e += 2) {
            const int d = r * 16 + e + lhalf * 8;
            *(unsigned*)&orow[d] = pack_bf16(O[i][e] * f, O[i][e + 1] * f);
        }
    }
}

// ---------------------------------------------------------------------------
// Kernel 3: y[b][d][t] = sum_k W[k][d] * out[b][k][t]  (K=512, bf16 WMMA)
// grid: (B, S/64), block 256
// ---------------------------------------------------------------------------
__global__ __launch_bounds__(256) void yproj_kernel(
    const unsigned short* __restrict__ Wt, const unsigned short* __restrict__ outT,
    float* __restrict__ y) {
    const int b     = blockIdx.x;
    const int t0    = blockIdx.y * 64;
    const int wave  = threadIdx.x >> 5;
    const int lane  = threadIdx.x & 31;
    const int l16   = lane & 15;
    const int lhalf = lane >> 4;
    const int r     = wave & 3;
    const int cb    = (wave >> 2) * 2;

    v8f C[2] = {v8f{0}, v8f{0}};
    const int d = r * 16 + l16;
    const unsigned short* arow  = Wt + (size_t)d * 512;
    const unsigned short* brow0 =
        outT + ((size_t)b * S_LEN + t0 + (cb + 0) * 16 + l16) * 512;
    const unsigned short* brow1 =
        outT + ((size_t)b * S_LEN + t0 + (cb + 1) * 16 + l16) * 512;

    for (int k0 = 0; k0 < 512; k0 += 32) {
        const unsigned short* pa = arow + k0 + lhalf * 16;
        v16bf A = load_bf16x16(pa, pa + 8);
        const unsigned short* pb0 = brow0 + k0 + lhalf * 16;
        v16bf B0 = load_bf16x16(pb0, pb0 + 8);
        C[0] = __builtin_amdgcn_wmma_f32_16x16x32_bf16(
            false, A, false, B0, (short)0, C[0], false, false);
        const unsigned short* pb1 = brow1 + k0 + lhalf * 16;
        v16bf B1 = load_bf16x16(pb1, pb1 + 8);
        C[1] = __builtin_amdgcn_wmma_f32_16x16x32_bf16(
            false, A, false, B1, (short)0, C[1], false, false);
    }

#pragma unroll
    for (int i = 0; i < 2; ++i) {
        const int t = t0 + (cb + i) * 16 + l16;
#pragma unroll
        for (int e = 0; e < 8; ++e) {
            const int dd = r * 16 + e + lhalf * 8;
            y[((size_t)b * DHEAD + dd) * S_LEN + t] = C[i][e];
        }
    }
}

// ---------------------------------------------------------------------------
extern "C" void kernel_launch(void* const* d_in, const int* in_sizes, int n_in,
                              void* d_out, int out_size, void* d_ws,
                              size_t ws_size, hipStream_t stream) {
    const float* x  = (const float*)d_in[0];
    const float* Wq = (const float*)d_in[1];
    const float* Wk = (const float*)d_in[2];
    const float* Wv = (const float*)d_in[3];
    const float* W  = (const float*)d_in[4];

    float* y  = (float*)d_out;                       // B*D*S      = 524288
    float* Qo = y + 524288;                          // B*H*D*S    = 4194304
    float* Ko = Qo + 4194304;
    float* Vo = Ko + 4194304;

    uint8_t* ws = (uint8_t*)d_ws;
    unsigned short* Qt = (unsigned short*)(ws);                 // 8 MiB
    unsigned short* Kt = (unsigned short*)(ws + 8388608);       // 8 MiB
    unsigned short* Vb = (unsigned short*)(ws + 16777216);      // 8 MiB
    unsigned short* oT = (unsigned short*)(ws + 25165824);      // 8 MiB
    unsigned short* Wt = (unsigned short*)(ws + 33554432);      // 64 KiB

    wtrans_kernel<<<16, 256, 0, stream>>>(W, Wt);
    qkv_proj_kernel<<<dim3(NBATCH * NHEAD, S_LEN / 128), 256, 0, stream>>>(
        x, Wq, Wk, Wv, Qo, Ko, Vo, Qt, Kt, Vb);
    attn_kernel<<<dim3(NBATCH * NHEAD, S_LEN / 64), 256, 0, stream>>>(
        Qt, Kt, Vb, oT);
    yproj_kernel<<<dim3(NBATCH, S_LEN / 64), 256, 0, stream>>>(Wt, oT, y);
}
